// SwitchTransformerGate_16544214024856
// MI455X (gfx1250) — matrix operand order repaired and checked by reading, not compiled, not run on hardware
//
#include <hip/hip_runtime.h>
#include <hip/hip_bf16.h>

// CDNA5 / gfx1250, wave32. Switch-Transformer gate:
//   logits = x[N,2048] @ W[64,2048]^T ; softmax/argmax per token; LB loss.
// Memory-bound: 134MB of x @ 23.3TB/s ~= 5.8us floor. GEMM on WMMA bf16.

typedef __attribute__((ext_vector_type(16))) __bf16 v16bf;
typedef __attribute__((ext_vector_type(8)))  float  v8f;

#define D_DIM 2048
#define E_NUM 64
#define TOK_PER_BLOCK 128   // 8 waves x 16 tokens

// ---------------- kernel 0: W f32 -> bf16 copy in workspace ----------------
__global__ __launch_bounds__(256) void gate_prep(const float* __restrict__ W,
                                                 __bf16* __restrict__ wbf, int n) {
  int gid = blockIdx.x * blockDim.x + threadIdx.x;
  if (gid < n) wbf[gid] = (__bf16)W[gid];
}

// ---------------- kernel 1: fused GEMM + softmax/argmax + block partials ----
__global__ __launch_bounds__(256) void gate_main(const float*  __restrict__ x,
                                                 const __bf16* __restrict__ wbf,
                                                 float* __restrict__ out_idx,
                                                 float* __restrict__ out_scr,
                                                 float* __restrict__ part) {
  __shared__ float logits[TOK_PER_BLOCK * 65];   // stride 65: conflict-free
  __shared__ float maxv[TOK_PER_BLOCK];
  __shared__ float rcp [TOK_PER_BLOCK];
  __shared__ int   cnts[E_NUM];
  __shared__ float bp  [E_NUM][4];

  const int tid  = threadIdx.x;
  const int lane = tid & 31;
  const int wave = tid >> 5;
  const int half = lane >> 4;     // 0: lanes 0-15, 1: lanes 16-31
  const int m    = lane & 15;     // token row (A) / expert column (B) in tile

  if (tid < E_NUM) cnts[tid] = 0;

  // ---- WMMA GEMM: 16 tokens x 64 experts per wave, K-loop step 32 ----
  const int tokBase = blockIdx.x * TOK_PER_BLOCK + wave * 16;
  const float* xrow = x + (size_t)(tokBase + m) * D_DIM;
  // A 16x32 layout (matches B): lanes0-15 hold K0..15, lanes16-31 K16..31.
  const int c1 = 16 * half;
  const int c2 = 16 * half + 8;
  // B 32x16 layout (contiguous, matches A): lanes0-15 hold K0..15, lanes16-31 K16..31.
  const int bofs = 16 * half;

  v8f acc0 = {}; v8f acc1 = {}; v8f acc2 = {}; v8f acc3 = {};

  for (int k0 = 0; k0 < D_DIM; k0 += 32) {
    // lanes L and L+16 together cover one full 128B cacheline of the row
    float4 f0 = *reinterpret_cast<const float4*>(xrow + k0 + c1);
    float4 f1 = *reinterpret_cast<const float4*>(xrow + k0 + c1 + 4);
    float4 f2 = *reinterpret_cast<const float4*>(xrow + k0 + c2);
    float4 f3 = *reinterpret_cast<const float4*>(xrow + k0 + c2 + 4);
    v16bf a;
    a[0] = (__bf16)f0.x; a[1] = (__bf16)f0.y; a[2] = (__bf16)f0.z; a[3] = (__bf16)f0.w;
    a[4] = (__bf16)f1.x; a[5] = (__bf16)f1.y; a[6] = (__bf16)f1.z; a[7] = (__bf16)f1.w;
    a[8] = (__bf16)f2.x; a[9] = (__bf16)f2.y; a[10]= (__bf16)f2.z; a[11]= (__bf16)f2.w;
    a[12]= (__bf16)f3.x; a[13]= (__bf16)f3.y; a[14]= (__bf16)f3.z; a[15]= (__bf16)f3.w;

    const __bf16* wb = wbf + k0 + bofs + (size_t)m * D_DIM;
    v16bf b0 = *reinterpret_cast<const v16bf*>(wb + (size_t)(0 * 16) * D_DIM);
    v16bf b1 = *reinterpret_cast<const v16bf*>(wb + (size_t)(1 * 16) * D_DIM);
    v16bf b2 = *reinterpret_cast<const v16bf*>(wb + (size_t)(2 * 16) * D_DIM);
    v16bf b3 = *reinterpret_cast<const v16bf*>(wb + (size_t)(3 * 16) * D_DIM);

    acc0 = __builtin_amdgcn_wmma_f32_16x16x32_bf16(false, a, false, b0, (short)0, acc0, false, false);
    acc1 = __builtin_amdgcn_wmma_f32_16x16x32_bf16(false, a, false, b1, (short)0, acc1, false, false);
    acc2 = __builtin_amdgcn_wmma_f32_16x16x32_bf16(false, a, false, b2, (short)0, acc2, false, false);
    acc3 = __builtin_amdgcn_wmma_f32_16x16x32_bf16(false, a, false, b3, (short)0, acc3, false, false);
  }

  // ---- spill logits to LDS (C layout: VGPR r -> token r / r+8, lane%16 -> expert) ----
#pragma unroll
  for (int r = 0; r < 8; ++r) {
    int tokLocal = wave * 16 + r + 8 * half;
    float* dst = &logits[tokLocal * 65 + m];
    dst[0]  = acc0[r];
    dst[16] = acc1[r];
    dst[32] = acc2[r];
    dst[48] = acc3[r];
  }
  __syncthreads();

  // ---- per-token softmax stats: threads 0..127 each own one token ----
  if (tid < TOK_PER_BLOCK) {
    const float* row = &logits[tid * 65];
    float mx = row[0]; int arg = 0;
#pragma unroll
    for (int e = 1; e < E_NUM; ++e) {       // strict > => first-max (jnp.argmax)
      float v = row[e];
      if (v > mx) { mx = v; arg = e; }
    }
    float s = 0.f;
#pragma unroll
    for (int e = 0; e < E_NUM; ++e) s += __expf(row[e] - mx);
    float r = 1.f / s;                       // max prob = exp(0)/sum
    maxv[tid] = mx; rcp[tid] = r;
    int g = blockIdx.x * TOK_PER_BLOCK + tid;
    out_idx[g] = (float)arg;
    out_scr[g] = r;
    atomicAdd(&cnts[arg], 1);                // int LDS atomic: deterministic
  }
  __syncthreads();

  // ---- per-expert prob sums: thread -> (expert e, token quarter q) ----
  {
    int e = tid & 63, q = tid >> 6;
    float p = 0.f;
#pragma unroll 4
    for (int j = 0; j < 32; ++j) {
      int t = q * 32 + j;
      p += __expf(logits[t * 65 + e] - maxv[t]) * rcp[t];
    }
    bp[e][q] = p;
  }
  __syncthreads();

  // ---- write per-block partials (no float global atomics -> deterministic) ----
  if (tid < E_NUM) {
    float s = bp[tid][0] + bp[tid][1] + bp[tid][2] + bp[tid][3];
    part[(size_t)blockIdx.x * 2 * E_NUM + tid]          = s;
    part[(size_t)blockIdx.x * 2 * E_NUM + E_NUM + tid]  = (float)cnts[tid];
  }
}

// ---------------- kernel 2: reduce partials -> load-balance loss ------------
__global__ __launch_bounds__(64) void gate_final(const float* __restrict__ part,
                                                 float* __restrict__ out_loss,
                                                 int nBlocks, float nTok) {
  __shared__ float prod[E_NUM];
  int e = threadIdx.x;
  float s = 0.f, c = 0.f;
  for (int b = 0; b < nBlocks; ++b) {        // fixed order: deterministic
    s += part[(size_t)b * 2 * E_NUM + e];
    c += part[(size_t)b * 2 * E_NUM + E_NUM + e];
  }
  prod[e] = s * c;
  __syncthreads();
  if (e == 0) {
    float t = 0.f;
    for (int i = 0; i < E_NUM; ++i) t += prod[i];
    *out_loss = (float)E_NUM * t / (nTok * nTok);
  }
}

// ---------------- launcher --------------------------------------------------
extern "C" void kernel_launch(void* const* d_in, const int* in_sizes, int n_in,
                              void* d_out, int out_size, void* d_ws, size_t ws_size,
                              hipStream_t stream) {
  const float* x = (const float*)d_in[0];   // [4,4096,2048] f32
  const float* W = (const float*)d_in[1];   // [64,2048] f32
  float* out = (float*)d_out;               // [N idx | N score | 1 loss]

  const int N = in_sizes[0] / D_DIM;        // 16384
  const int nW = E_NUM * D_DIM;             // 131072
  const int nBlocks = N / TOK_PER_BLOCK;    // 128

  __bf16* wbf = (__bf16*)d_ws;                                   // 256 KB
  float*  part = (float*)((char*)d_ws + (size_t)nW * sizeof(__bf16)); // 64 KB

  gate_prep<<<(nW + 255) / 256, 256, 0, stream>>>(W, wbf, nW);
  gate_main<<<nBlocks, 256, 0, stream>>>(x, wbf, out, out + N, part);
  gate_final<<<1, 64, 0, stream>>>(part, out + 2 * N, nBlocks, (float)N);
}
